// LISTAEncoder_41205916238338
// MI455X (gfx1250) — matrix operand order repaired and checked by reading, not compiled, not run on hardware
//
#include <hip/hip_runtime.h>

typedef __attribute__((ext_vector_type(16))) __bf16 v16bf;
typedef __attribute__((ext_vector_type(8)))  __bf16 v8bf;
typedef __attribute__((ext_vector_type(8)))  float  v8f;
typedef __attribute__((ext_vector_type(4)))  float  v4f;
typedef __attribute__((ext_vector_type(4)))  unsigned int v4u;
typedef __attribute__((ext_vector_type(4)))  int    v4i;
typedef __attribute__((ext_vector_type(8)))  int    v8i;

#define N_ROWS    262144          // B*N = 16*16384
#define D_IN      128
#define D_LAT     256
#define T_ITER    8
#define WAVES     4               // waves per block
#define BLOCK_THREADS (WAVES * 32)
#define RPB       (WAVES * 16)    // rows per block = 64
#define MAIN_BLOCKS (N_ROWS / RPB)            // 4096
#define PREP_ELEMS  (T_ITER * D_LAT * D_LAT)  // 524288
#define PREP_BLOCKS (PREP_ELEMS / 256)        // 2048
#define OUT_ELEMS   (262144ull * 256ull)      // 67108864

static __device__ __forceinline__ v8f wmma_bf16(v16bf a, v16bf b, v8f c) {
  // D = A(16x32 bf16) * B(32x16 bf16) + C(16x16 f32)
  return __builtin_amdgcn_wmma_f32_16x16x32_bf16(false, a, false, b, (short)0, c,
                                                 false, false);
}

static __device__ __forceinline__ v16bf cat8(v8bf a, v8bf b) {
  return __builtin_shufflevector(a, b, 0, 1, 2, 3, 4, 5, 6, 7,
                                 8, 9, 10, 11, 12, 13, 14, 15);
}

// ---------------------------------------------------------------------------
// TDM: async DMA of one B k-panel (2D tile: 32 bf16 wide x 256 rows, row
// stride 256 elements) from global into LDS, packed as [n][32] contiguous.
// Descriptor layout per CDNA5 ISA (D# groups 0/1; 2D tensor,
// groups 2/3 zero). Tracked with TENSORcnt.
// ---------------------------------------------------------------------------
static __device__ __forceinline__ void tdm_panel_load(const __bf16* gsrc,
                                                      __bf16* ldst) {
  const unsigned long long ga = (unsigned long long)(uintptr_t)gsrc;
  const unsigned int lds = (unsigned int)(uintptr_t)ldst;  // flat addr[31:0] == LDS offset
  v4u g0;
  g0[0] = 1u;                                           // count=1, user descriptor
  g0[1] = lds;                                          // lds_addr
  g0[2] = (unsigned int)(ga & 0xffffffffu);             // global_addr[31:0]
  g0[3] = (unsigned int)((ga >> 32) & 0x01ffffffu) | (2u << 30);  // addr[56:32] | type=2
  v8i g1;
  g1[0] = (int)(1u << 16);      // wg_mask=0 | data_size=1 (2 bytes) | no flags
  g1[1] = (int)(32u << 16);     // atomic_barrier_addr=0 | tensor_dim0[15:0]=32
  g1[2] = (int)(256u << 16);    // tensor_dim0[31:16]=0 | tensor_dim1[15:0]=256
  g1[3] = (int)(32u << 16);     // tensor_dim1[31:16]=0 | tile_dim0=32
  g1[4] = 256;                  // tile_dim1=256 | tile_dim2=0
  g1[5] = 256;                  // tensor_dim0_stride[31:0]=256
  g1[6] = 0;                    // stride0[47:32]=0 | tensor_dim1_stride[15:0]=0
  g1[7] = 0;                    // tensor_dim1_stride[47:16]=0
  const v4i z4 = {0, 0, 0, 0};
#if defined(__clang_major__) && (__clang_major__ >= 23)
  const v8i z8 = {0, 0, 0, 0, 0, 0, 0, 0};
  __builtin_amdgcn_tensor_load_to_lds(g0, g1, z4, z4, z8, 0);
#else
  __builtin_amdgcn_tensor_load_to_lds(g0, g1, z4, z4, 0);
#endif
}

// ---------------------------------------------------------------------------
// Prep: split We into hi/lo bf16, and build B_t = I - step_t * S as hi/lo bf16.
//   Bstore[t][n][k] = (n==k) - step_t * S[n][k]
//   Westore[n][k]   = We[n][k]
// ---------------------------------------------------------------------------
__global__ void lista_prep(const float* __restrict__ We,
                           const float* __restrict__ S,
                           const float* __restrict__ steps,
                           __bf16* __restrict__ We_hi, __bf16* __restrict__ We_lo,
                           __bf16* __restrict__ B_hi,  __bf16* __restrict__ B_lo) {
  const int idx = blockIdx.x * 256 + threadIdx.x;
  if (idx < PREP_ELEMS) {
    const int t   = idx >> 16;
    const int rem = idx & 65535;
    const int n   = rem >> 8;
    const int k   = rem & 255;
    const float v = ((n == k) ? 1.0f : 0.0f) - steps[t] * S[n * D_LAT + k];
    const __bf16 h = (__bf16)v;
    B_hi[idx] = h;
    B_lo[idx] = (__bf16)(v - (float)h);
  }
  if (idx < D_LAT * D_IN) {
    const float v = We[idx];
    const __bf16 h = (__bf16)v;
    We_hi[idx] = h;
    We_lo[idx] = (__bf16)(v - (float)h);
  }
}

// ---------------------------------------------------------------------------
// Main fused kernel: encoder GEMM + 8 LISTA iterations + threshold + output.
// Each wave owns 16 rows; z and b stay in VGPR f32 accumulators throughout.
// B panels are DMAed into LDS by the Tensor Data Mover, double buffered.
// ---------------------------------------------------------------------------
__global__ __launch_bounds__(BLOCK_THREADS, 2)
void lista_main(const float* __restrict__ x,
                const float* __restrict__ steps,
                const float* __restrict__ thresholds,
                const __bf16* __restrict__ We_hi, const __bf16* __restrict__ We_lo,
                const __bf16* __restrict__ B_hi,  const __bf16* __restrict__ B_lo,
                float* __restrict__ out,
                float* __restrict__ partials) {
  // double-buffered B k-panels (32 K x 256 cols, hi+lo): 2*2*16KB = 64KB
  __shared__ __align__(64) __bf16 sPanel[2][2][D_LAT * 32];
  // per-wave z relayout scratch (hi+lo bf16 of 16x256): 4 * 16KB = 64KB
  __shared__ __align__(64) __bf16 sZ[WAVES][2][16 * D_LAT];
  __shared__ float sRed[WAVES];

  const int tid  = threadIdx.x;
  const int w    = tid >> 5;
  const int lane = tid & 31;
  const int lr   = lane & 15;   // lane % 16
  const int lhi  = lane >> 4;   // lane half
  const int row0 = blockIdx.x * RPB + w * 16;

  v8f acc[16];    // current z tile, f32, D-layout (16 rows x 256 cols)
  v8f bsave[16];  // encoder output b, kept for C-init each iteration

  // ---- kick off first B panel DMA (overlaps with encoder GEMM) ----
  if (w == 0) {
    tdm_panel_load(B_hi, &sPanel[0][0][0]);
    tdm_panel_load(B_lo, &sPanel[0][1][0]);
  }

  // ---- Phase A: b = x @ We^T, split-bf16, f32 accumulate ----
  #pragma unroll
  for (int nb = 0; nb < 16; ++nb) {
    #pragma unroll
    for (int e = 0; e < 8; ++e) acc[nb][e] = 0.0f;
  }

  const float* xrow = x + (size_t)(row0 + lr) * D_IN;
  #pragma unroll
  for (int kk = 0; kk < 4; ++kk) {
    // A operand (16x32 bf16 layout): lane holds row lr; K = kk*32 + lhi*16 + {0..15}
    const int kb = kk * 32 + lhi * 16;
    v4f fq[4];
    fq[0] = *(const v4f*)(xrow + kb);
    fq[1] = *(const v4f*)(xrow + kb + 4);
    fq[2] = *(const v4f*)(xrow + kb + 8);
    fq[3] = *(const v4f*)(xrow + kb + 12);
    v16bf ahi, alo;
    #pragma unroll
    for (int i = 0; i < 16; ++i) {
      const float v = fq[i >> 2][i & 3];
      const __bf16 h = (__bf16)v;
      ahi[i] = h;
      alo[i] = (__bf16)(v - (float)h);
    }
    const int ks = kk * 32 + lhi * 16;  // B operand: lanes 0-15 K 0..15, lanes 16-31 K 16..31
    #pragma unroll
    for (int nb = 0; nb < 16; ++nb) {
      const int n = nb * 16 + lr;
      const v16bf bh = *(const v16bf*)(We_hi + n * D_IN + ks);
      const v16bf bl = *(const v16bf*)(We_lo + n * D_IN + ks);
      acc[nb] = wmma_bf16(ahi, bh, acc[nb]);
      acc[nb] = wmma_bf16(alo, bh, acc[nb]);
      acc[nb] = wmma_bf16(ahi, bl, acc[nb]);
    }
  }
  #pragma unroll
  for (int nb = 0; nb < 16; ++nb) bsave[nb] = acc[nb];

  __bf16* zh = &sZ[w][0][0];
  __bf16* zl = &sZ[w][1][0];
  int buf = 0;
  float psum = 0.0f;

  // ---- Phase B: 8 LISTA iterations: z = soft( z*(I-aS)^T + a*b, th ) ----
  for (int t = 0; t < T_ITER; ++t) {
    const float st = steps[t];
    const float th = fabsf(thresholds[t]);

    // spill z_in (acc) into per-wave LDS as hi/lo bf16; reinit acc = st*b
    #pragma unroll
    for (int nb = 0; nb < 16; ++nb) {
      const int l = nb * 16 + lr;
      #pragma unroll
      for (int e = 0; e < 8; ++e) {
        const int r = e + lhi * 8;          // D-layout row
        const float v = acc[nb][e];
        const __bf16 h = (__bf16)v;
        zh[r * D_LAT + l] = h;
        zl[r * D_LAT + l] = (__bf16)(v - (float)h);
      }
      acc[nb] = bsave[nb] * st;
    }

    for (int kk = 0; kk < 8; ++kk) {
      // wave 0: DMA filling sPanel[buf] (issued last step) must be complete
      if (w == 0) __builtin_amdgcn_s_wait_tensorcnt(0);
      __syncthreads();  // publish sPanel[buf]; everyone done reading sPanel[buf^1]
      const bool last = (t == T_ITER - 1) && (kk == 7);
      if (!last && w == 0) {
        int nk = kk + 1, nt = t;
        if (nk == 8) { nk = 0; nt = t + 1; }
        const size_t base = ((size_t)nt << 16) + (size_t)nk * 32;
        tdm_panel_load(B_hi + base, &sPanel[buf ^ 1][0][0]);
        tdm_panel_load(B_lo + base, &sPanel[buf ^ 1][1][0]);
      }
      // A operand from z scratch (row-major, contiguous per lane)
      const int ka = kk * 32 + lhi * 16;
      const v16bf ahi = cat8(*(const v8bf*)&zh[lr * D_LAT + ka],
                             *(const v8bf*)&zh[lr * D_LAT + ka + 8]);
      const v16bf alo = cat8(*(const v8bf*)&zl[lr * D_LAT + ka],
                             *(const v8bf*)&zl[lr * D_LAT + ka + 8]);
      const __bf16* sph = &sPanel[buf][0][0];
      const __bf16* spl = &sPanel[buf][1][0];
      #pragma unroll
      for (int nb = 0; nb < 16; ++nb) {
        const int n = nb * 16 + lr;
        const v16bf bh = *(const v16bf*)(sph + n * 32 + lhi * 16);
        const v16bf bl = *(const v16bf*)(spl + n * 32 + lhi * 16);
        acc[nb] = wmma_bf16(ahi, bh, acc[nb]);
        acc[nb] = wmma_bf16(alo, bh, acc[nb]);
        acc[nb] = wmma_bf16(ahi, bl, acc[nb]);
      }
      buf ^= 1;
    }

    // soft threshold: z = sign(z)*max(|z|-th, 0)
    const bool final_t = (t == T_ITER - 1);
    #pragma unroll
    for (int nb = 0; nb < 16; ++nb) {
      #pragma unroll
      for (int e = 0; e < 8; ++e) {
        const float v = acc[nb][e];
        const float a = fmaxf(fabsf(v) - th, 0.0f);
        acc[nb][e] = (v >= 0.0f) ? a : -a;
        if (final_t) psum += a;
      }
    }
  }

  // ---- output: stage f32 z through LDS (reuse sZ[w], 16KB), coalesced stores ----
  float* sZf = (float*)zh;
  #pragma unroll
  for (int nb = 0; nb < 16; ++nb) {
    const int l = nb * 16 + lr;
    #pragma unroll
    for (int e = 0; e < 8; ++e) sZf[(e + lhi * 8) * D_LAT + l] = acc[nb][e];
  }
  for (int rr = 0; rr < 16; ++rr) {
    const size_t row = (size_t)(row0 + rr);
    v4f* dst = (v4f*)(out + row * D_LAT + lane * 8);
    const v4f* src = (const v4f*)(sZf + rr * D_LAT + lane * 8);
    dst[0] = src[0];
    dst[1] = src[1];
  }

  // ---- deterministic block-level |z| partial sum ----
  #pragma unroll
  for (int off = 16; off > 0; off >>= 1) psum += __shfl_down(psum, off, 32);
  if (lane == 0) sRed[w] = psum;
  __syncthreads();
  if (tid == 0) {
    float s = 0.0f;
    for (int i = 0; i < WAVES; ++i) s += sRed[i];
    partials[blockIdx.x] = s;
  }
}

__global__ void lista_reduce(const float* __restrict__ partials,
                             float* __restrict__ out_scalar) {
  __shared__ float red[256];
  float s = 0.0f;
  for (int i = threadIdx.x; i < MAIN_BLOCKS; i += 256) s += partials[i];
  red[threadIdx.x] = s;
  __syncthreads();
  for (int stride = 128; stride > 0; stride >>= 1) {
    if (threadIdx.x < stride) red[threadIdx.x] += red[threadIdx.x + stride];
    __syncthreads();
  }
  if (threadIdx.x == 0) out_scalar[0] = red[0] * (1.0f / (float)OUT_ELEMS);
}

extern "C" void kernel_launch(void* const* d_in, const int* in_sizes, int n_in,
                              void* d_out, int out_size, void* d_ws, size_t ws_size,
                              hipStream_t stream) {
  const float* x     = (const float*)d_in[0];
  const float* We    = (const float*)d_in[1];
  const float* S     = (const float*)d_in[2];
  const float* steps = (const float*)d_in[3];
  const float* thr   = (const float*)d_in[4];
  float* out = (float*)d_out;

  char* ws = (char*)d_ws;
  __bf16* We_hi = (__bf16*)ws;             ws += (size_t)D_LAT * D_IN * 2;
  __bf16* We_lo = (__bf16*)ws;             ws += (size_t)D_LAT * D_IN * 2;
  __bf16* B_hi  = (__bf16*)ws;             ws += (size_t)PREP_ELEMS * 2;
  __bf16* B_lo  = (__bf16*)ws;             ws += (size_t)PREP_ELEMS * 2;
  float* partials = (float*)ws;            // MAIN_BLOCKS floats; total ~2.25MB

  lista_prep<<<PREP_BLOCKS, 256, 0, stream>>>(We, S, steps, We_hi, We_lo, B_hi, B_lo);
  lista_main<<<MAIN_BLOCKS, BLOCK_THREADS, 0, stream>>>(
      x, steps, thr, We_hi, We_lo, B_hi, B_lo, out, partials);
  lista_reduce<<<1, 256, 0, stream>>>(partials, out + OUT_ELEMS);
}
